// AttentionReadout_18743237280334
// MI455X (gfx1250) — matrix operand order repaired and checked by reading, not compiled, not run on hardware
//
#include <hip/hip_runtime.h>
#include <hip/hip_bf16.h>

#define H 512
#define NH 8
#define HD 64

typedef __bf16  v16bf __attribute__((ext_vector_type(16)));
typedef float   v8f   __attribute__((ext_vector_type(8)));

// native f32 -> bf16 (RNE) — lets clang use v_cvt_pk_bf16_f32
static __device__ __forceinline__ __bf16 f2bf(float f) { return (__bf16)f; }

// load 8 consecutive floats as two b128s and convert into frag[base..base+7]
static __device__ __forceinline__ void load_bf8(v16bf& frag, int base, const float* __restrict__ p) {
    float4 a = *(const float4*)(p + 0);
    float4 b = *(const float4*)(p + 4);
    frag[base + 0] = f2bf(a.x); frag[base + 1] = f2bf(a.y);
    frag[base + 2] = f2bf(a.z); frag[base + 3] = f2bf(a.w);
    frag[base + 4] = f2bf(b.x); frag[base + 5] = f2bf(b.y);
    frag[base + 6] = f2bf(b.z); frag[base + 7] = f2bf(b.w);
}

// orderable encoding for float max via unsigned atomicMax
static __device__ __forceinline__ unsigned fenc(float f) {
    unsigned u = __float_as_uint(f);
    return (u & 0x80000000u) ? ~u : (u | 0x80000000u);
}
static __device__ __forceinline__ float fdec(unsigned u) {
    unsigned v = (u & 0x80000000u) ? (u & 0x7FFFFFFFu) : ~u;
    return __uint_as_float(v);
}

static __device__ __forceinline__ int lower_bound_i(const int* b, int n, int key) {
    int lo = 0, hi = n;
    while (lo < hi) { int mid = (lo + hi) >> 1; if (b[mid] < key) lo = mid + 1; else hi = mid; }
    return lo;
}

// ---------------------------------------------------------------------------
// Kernel 1: fold query into k_w / k_b:  A[j*8+h] = sum_d k_w[(h*64+d)*512+j]*q[h*64+d]/8
// ---------------------------------------------------------------------------
__global__ void k_prep(const float* __restrict__ k_w, const float* __restrict__ k_b,
                       const float* __restrict__ query, float* __restrict__ Amat,
                       float* __restrict__ cvec) {
    const float scale = 0.125f;  // 1/sqrt(64)
    int t = blockIdx.x * blockDim.x + threadIdx.x;
    if (t < H * NH) {
        int j = t >> 3, h = t & 7;
        float acc = 0.f;
        #pragma unroll 4
        for (int d = 0; d < HD; ++d)
            acc += k_w[(size_t)(h * HD + d) * H + j] * query[h * HD + d];
        Amat[j * NH + h] = acc * scale;
    }
    if (t < NH) {
        float acc = 0.f;
        for (int d = 0; d < HD; ++d) acc += k_b[t * HD + d] * query[t * HD + d];
        cvec[t] = acc * scale;
    }
}

// ---------------------------------------------------------------------------
// Kernel 2: logits  s[n,h] = x[n,:] . A[:,h] + c[h]   (one wave32 per node)
// ---------------------------------------------------------------------------
__global__ void k_logits(const float* __restrict__ x, const float* __restrict__ Amat,
                         const float* __restrict__ cvec, float* __restrict__ s, int N) {
    __shared__ float A_lds[H * NH];
    __shared__ float c_lds[NH];
    for (int i = threadIdx.x; i < H * NH; i += blockDim.x) A_lds[i] = Amat[i];
    if (threadIdx.x < NH) c_lds[threadIdx.x] = cvec[threadIdx.x];
    __syncthreads();

    int wave = threadIdx.x >> 5;
    int lane = threadIdx.x & 31;
    int n = blockIdx.x * 8 + wave;
    if (n >= N) return;

    const float* xr = x + (size_t)n * H;
    float acc[NH];
    #pragma unroll
    for (int h = 0; h < NH; ++h) acc[h] = 0.f;
    #pragma unroll
    for (int i = 0; i < H / 32; ++i) {
        int j = lane + i * 32;
        float xv = xr[j];
        const float* Ar = &A_lds[j * NH];
        #pragma unroll
        for (int h = 0; h < NH; ++h) acc[h] += xv * Ar[h];
    }
    #pragma unroll
    for (int h = 0; h < NH; ++h)
        for (int off = 16; off; off >>= 1) acc[h] += __shfl_xor(acc[h], off, 32);
    if (lane == 0) {
        #pragma unroll
        for (int h = 0; h < NH; ++h) s[(size_t)n * NH + h] = acc[h] + c_lds[h];
    }
}

// ---------------------------------------------------------------------------
// Kernel 3: per-graph segment softmax + weighted x reduction
//   Xw[g,h,:] = sum_{n in g} exp(s[n,h]-max) * x[n,:],  denom[g,h] = sum exp(...)
// One block (512 threads) per graph; batch is sorted.
// ---------------------------------------------------------------------------
__global__ void k_segment(const float* __restrict__ x, const int* __restrict__ batch,
                          const float* __restrict__ s, float* __restrict__ Xw,
                          float* __restrict__ denom_out, int N) {
    __shared__ int sh_start, sh_end;
    __shared__ unsigned sh_maxenc[NH];
    __shared__ float sh_max[NH];
    __shared__ float sh_den[NH];
    __shared__ float wbuf[64 * NH];

    int g = blockIdx.x;
    int t = threadIdx.x;
    if (t == 0) { sh_start = lower_bound_i(batch, N, g); sh_end = lower_bound_i(batch, N, g + 1); }
    if (t < NH) { sh_maxenc[t] = 0u; sh_den[t] = 0.f; }
    __syncthreads();
    int start = sh_start, end = sh_end;

    // ---- phase A: per-head segment max (order-independent -> deterministic)
    float lmax[NH];
    #pragma unroll
    for (int h = 0; h < NH; ++h) lmax[h] = -__builtin_inff();
    for (int n = start + t; n < end; n += blockDim.x) {
        #pragma unroll
        for (int h = 0; h < NH; ++h) lmax[h] = fmaxf(lmax[h], s[(size_t)n * NH + h]);
    }
    #pragma unroll
    for (int h = 0; h < NH; ++h) atomicMax(&sh_maxenc[h], fenc(lmax[h]));
    __syncthreads();
    if (t < NH) sh_max[t] = fdec(sh_maxenc[t]);
    __syncthreads();

    // ---- phase B: chunked exp weights + weighted column accumulation
    float accXw[NH];
    #pragma unroll
    for (int h = 0; h < NH; ++h) accXw[h] = 0.f;

    for (int base = start; base < end; base += 64) {
        int cnt = min(64, end - base);
        if (base + 64 < end)
            __builtin_prefetch(&x[(size_t)(base + 64) * H + t], 0, 0);  // global_prefetch_b8
        if (t < cnt * NH) {
            int i = t >> 3, h = t & 7;
            wbuf[t] = __expf(s[(size_t)(base + i) * NH + h] - sh_max[h]);
        }
        __syncthreads();
        if (t < NH) {               // deterministic serial denom per head
            float d = 0.f;
            for (int i = 0; i < cnt; ++i) d += wbuf[i * NH + t];
            sh_den[t] += d;
        }
        for (int i = 0; i < cnt; ++i) {
            float xv = x[(size_t)(base + i) * H + t];   // coalesced row read
            #pragma unroll
            for (int h = 0; h < NH; ++h) accXw[h] += wbuf[i * NH + h] * xv;
        }
        __syncthreads();
    }
    #pragma unroll
    for (int h = 0; h < NH; ++h) Xw[(size_t)g * (NH * H) + h * H + t] = accXw[h];
    if (t < NH) denom_out[g * NH + t] = sh_den[t];
}

// ---------------------------------------------------------------------------
// WMMA fragment index helpers (16-bit A 16x32, B 32x16, f32 C/D 16x16)
//   A: lane L -> M = L&15; elems 0..7 -> K = (L&16?8:0)+e ; elems 8..15 -> K = +16
//   B: lane L -> N = L&15; elems 0..7 -> K = (L&16?8:0)+e ; elems 8..15 -> K = +16
//   D: vgpr r -> M = r + (L&16?8:0); N = L&15
// ---------------------------------------------------------------------------

// Kernel 4: numer[g,h,d] = Xw[g,h,:] . v_w[h*64+d,:]  (+ denom*v_b), then
// attended = numer/denom_safe.   bf16 WMMA, one block = (head, 16-graph tile).
__global__ void k_numer_wmma(const float* __restrict__ Xw, const float* __restrict__ denom,
                             const float* __restrict__ v_w, const float* __restrict__ v_b,
                             float* __restrict__ att, int G) {
    int wave = threadIdx.x >> 5;
    int lane = threadIdx.x & 31;
    int mtiles = G >> 4;
    int h  = blockIdx.x / mtiles;
    int mt = blockIdx.x % mtiles;
    int m0 = mt * 16;
    int d0 = wave * 16;            // 4 waves cover HD=64
    int Lhi  = lane & 16;
    int mrow = lane & 15;
    int ab = Lhi ? 8 : 0;
    int bb = Lhi ? 8 : 0;

    const float* Abase = Xw + (size_t)(m0 + mrow) * (NH * H) + h * H;
    const float* Bbase = v_w + (size_t)(h * HD + d0 + mrow) * H;

    v8f acc = {};
    #pragma unroll 4
    for (int kk = 0; kk < H; kk += 32) {
        v16bf a, b;
        load_bf8(a, 0, Abase + kk + ab);
        load_bf8(a, 8, Abase + kk + ab + 16);
        load_bf8(b, 0, Bbase + kk + bb);
        load_bf8(b, 8, Bbase + kk + bb + 16);
        acc = __builtin_amdgcn_wmma_f32_16x16x32_bf16(false, a, false, b, (short)0, acc,
                                                      false, false);
    }
    int ncol = lane & 15;
    int d = d0 + ncol;
    float vb = v_b[h * HD + d];
    #pragma unroll
    for (int r = 0; r < 8; ++r) {
        int g = m0 + r + (Lhi ? 8 : 0);
        float den  = denom[g * NH + h];
        float inv  = (den > 0.f) ? __builtin_amdgcn_rcpf(den) : 1.f;
        float val  = acc[r] * inv + (den > 0.f ? vb : 0.f);
        att[(size_t)g * H + h * HD + d] = val;
    }
}

// Kernel 5: proj[g,i] = attended[g,:] . o_w[i,:] + o_b[i]   (bf16 WMMA GEMM)
__global__ void k_proj_wmma(const float* __restrict__ att, const float* __restrict__ o_w,
                            const float* __restrict__ o_b, float* __restrict__ proj, int G) {
    int wave = threadIdx.x >> 5;
    int lane = threadIdx.x & 31;
    int tile = blockIdx.x * 4 + wave;
    int ntiles = H / 16;           // 32
    int mt = tile / ntiles, nt = tile % ntiles;
    int m0 = mt * 16, n0 = nt * 16;
    int Lhi  = lane & 16;
    int mrow = lane & 15;
    int ab = Lhi ? 8 : 0;
    int bb = Lhi ? 8 : 0;

    const float* Abase = att + (size_t)(m0 + mrow) * H;
    const float* Bbase = o_w + (size_t)(n0 + mrow) * H;

    v8f acc = {};
    #pragma unroll 4
    for (int kk = 0; kk < H; kk += 32) {
        v16bf a, b;
        load_bf8(a, 0, Abase + kk + ab);
        load_bf8(a, 8, Abase + kk + ab + 16);
        load_bf8(b, 0, Bbase + kk + bb);
        load_bf8(b, 8, Bbase + kk + bb + 16);
        acc = __builtin_amdgcn_wmma_f32_16x16x32_bf16(false, a, false, b, (short)0, acc,
                                                      false, false);
    }
    int ncol = lane & 15;
    float bias = o_b[n0 + ncol];
    #pragma unroll
    for (int r = 0; r < 8; ++r) {
        int m = m0 + r + (Lhi ? 8 : 0);
        proj[(size_t)m * H + n0 + ncol] = acc[r] + bias;
    }
}

// Kernel 6: LayerNorm over each output row (512 threads per row)
__global__ void k_layernorm(const float* __restrict__ proj, const float* __restrict__ ln_g,
                            const float* __restrict__ ln_b, float* __restrict__ out) {
    __shared__ float rs[16], rs2[16];
    int g = blockIdx.x, t = threadIdx.x;
    float v = proj[(size_t)g * H + t];
    float sum = v, sq = v * v;
    int lane = t & 31, wave = t >> 5;
    for (int off = 16; off; off >>= 1) { sum += __shfl_xor(sum, off, 32); sq += __shfl_xor(sq, off, 32); }
    if (lane == 0) { rs[wave] = sum; rs2[wave] = sq; }
    __syncthreads();
    if (t == 0) {
        float a = 0.f, b = 0.f;
        for (int i = 0; i < 16; ++i) { a += rs[i]; b += rs2[i]; }
        rs[0] = a; rs2[0] = b;
    }
    __syncthreads();
    float mean = rs[0] / (float)H;
    float var  = rs2[0] / (float)H - mean * mean;
    float inv  = rsqrtf(var + 1e-5f);
    out[(size_t)g * H + t] = (v - mean) * inv * ln_g[t] + ln_b[t];
}

// ---------------------------------------------------------------------------
extern "C" void kernel_launch(void* const* d_in, const int* in_sizes, int n_in,
                              void* d_out, int out_size, void* d_ws, size_t ws_size,
                              hipStream_t stream) {
    const float* x     = (const float*)d_in[0];
    const int*   batch = (const int*)  d_in[1];
    const float* query = (const float*)d_in[2];
    const float* k_w   = (const float*)d_in[3];
    const float* k_b   = (const float*)d_in[4];
    const float* v_w   = (const float*)d_in[5];
    const float* v_b   = (const float*)d_in[6];
    const float* o_w   = (const float*)d_in[7];
    const float* o_b   = (const float*)d_in[8];
    const float* ln_g  = (const float*)d_in[9];
    const float* ln_b  = (const float*)d_in[10];
    float* out = (float*)d_out;

    int N = in_sizes[0] / H;
    int G = out_size / H;          // 256

    // workspace carve-up (floats)
    float* ws    = (float*)d_ws;
    float* Amat  = ws;                                   // 4096
    float* cvec  = Amat + H * NH;                        // 8
    float* s     = cvec + NH;                            // N*8
    float* Xw    = s + (size_t)N * NH;                   // G*4096
    float* denom = Xw + (size_t)G * NH * H;              // G*8
    float* att   = denom + (size_t)G * NH;               // G*512
    float* proj  = att + (size_t)G * H;                  // G*512

    k_prep<<<(H * NH + 255) / 256, 256, 0, stream>>>(k_w, k_b, query, Amat, cvec);
    k_logits<<<(N + 7) / 8, 256, 0, stream>>>(x, Amat, cvec, s, N);
    k_segment<<<G, H, 0, stream>>>(x, batch, s, Xw, denom, N);
    k_numer_wmma<<<NH * (G / 16), 128, 0, stream>>>(Xw, denom, v_w, v_b, att, G);
    int proj_blocks = ((G / 16) * (H / 16)) / 4;
    k_proj_wmma<<<proj_blocks, 128, 0, stream>>>(att, o_w, o_b, proj, G);
    k_layernorm<<<G, H, 0, stream>>>(proj, ln_g, ln_b, out);
}
